// SelectSubAttention_5403068858798
// MI455X (gfx1250) — matrix operand order repaired and checked by reading, not compiled, not run on hardware
//
#include <hip/hip_runtime.h>
#include <hip/hip_bf16.h>

// ---------------- problem constants (from reference) ----------------
#define BB   128          // batch
#define NN   256          // nodes per graph
#define DD   256          // model dim
#define HH   8            // heads
#define HD   32           // head dim
#define KSEL 128          // kept nodes per graph
#define EE   524288       // total edges
#define SENT 16384        // B*KSEL sentinel
#define MTOT (BB*NN)      // 32768 rows

typedef __bf16 bf16_t;
typedef __attribute__((ext_vector_type(16))) __bf16 v16bf;
typedef __attribute__((ext_vector_type(8)))  __bf16 v8bf;
typedef __attribute__((ext_vector_type(8)))  float  v8f;
typedef __attribute__((ext_vector_type(4)))  unsigned u32x4;
typedef __attribute__((ext_vector_type(8)))  int      i32x8;
typedef __attribute__((ext_vector_type(4)))  int      i32x4;

#if defined(__has_builtin)
#if __has_builtin(__builtin_amdgcn_tensor_load_to_lds) && __has_builtin(__builtin_amdgcn_s_wait_tensorcnt)
#define HAVE_TDM 1
#endif
#endif

// ---------------- CDNA5 async global->LDS (ASYNCcnt) ----------------
__device__ inline void async_ld_b128(unsigned lds_off, const void* gaddr) {
  asm volatile("global_load_async_to_lds_b128 %0, %1, off"
               :: "v"(lds_off), "v"((unsigned long long)(uintptr_t)gaddr)
               : "memory");
}
__device__ inline void wait_async0() {
  asm volatile("s_wait_asynccnt 0" ::: "memory");
}

#if HAVE_TDM
// Issue one TDM 2D tile load: 128 rows x 32 cols of bf16 into LDS at ldsOff.
__device__ inline void tdm_issue_tile(const bf16_t* A, int rowBase, int kkElem,
                                      unsigned ldsOff) {
  unsigned long long ga =
      (unsigned long long)(uintptr_t)(A + (size_t)rowBase * DD + kkElem);
  u32x4 g0 = { 1u,                                    // count=1, user mode
               ldsOff,                                // lds_addr
               (unsigned)(ga & 0xffffffffu),
               (unsigned)((ga >> 32) & 0x01ffffffu) | (2u << 30) }; // type=2
  i32x8 g1 = { (int)(1u << 16),                                  // data_size=1 (2B)
               (int)(((unsigned)DD & 0xffffu) << 16),            // tensor_dim0 lo16
               (int)((((unsigned)DD >> 16) & 0xffffu) |
                     (((unsigned)MTOT & 0xffffu) << 16)),        // dim0 hi | dim1 lo
               (int)((((unsigned)MTOT >> 16) & 0xffffu) |
                     (32u << 16)),                               // dim1 hi | tile_dim0
               (int)128u,                                        // tile_dim1 (tile_dim2=0)
               (int)(unsigned)DD,                                // dim0_stride lo32
               0, 0 };
  i32x4 g2 = {0, 0, 0, 0};
  i32x4 g3 = {0, 0, 0, 0};
  i32x8 g4 = {0, 0, 0, 0, 0, 0, 0, 0};
  __builtin_amdgcn_tensor_load_to_lds(g0, g1, g2, g3, g4, 0);
}
#endif

// ---------------- WMMA fragment loaders (CDNA5 wave32 layouts) ----------------
// A: 16x32 bf16 row-major tile; lane l: row = l&15, k = (l>>4)*16 + j (contiguous)
__device__ inline v16bf load_a_frag(const bf16_t* p0, int row_stride) {
  int lane = threadIdx.x & 31;
  const bf16_t* p = p0 + (lane & 15) * row_stride + (lane >> 4) * 16;
  return *(const v16bf*)(p);          // K kh..kh+15
}
// B from pre-packed per-lane fragment layout: lane's 16 bf16 contiguous at lane*16
__device__ inline v16bf load_bfrag_packed(const bf16_t* base) {
  return *(const v16bf*)(base + (threadIdx.x & 31) * 16);
}
// B = T^T where T is a row-major [n][k] tile: element j = T[col][kb+j] -> contiguous
__device__ inline v16bf load_bt_frag(const bf16_t* lds, int k_stride) {
  int lane = threadIdx.x & 31;
  return *(const v16bf*)(lds + (lane & 15) * k_stride + (lane >> 4) * 16);
}

// ---------------- f32 -> bf16 convert ----------------
__global__ __launch_bounds__(256) void cvt_k(const float* __restrict__ src,
                                             bf16_t* __restrict__ dst, int n) {
  int i = blockIdx.x * 256 + threadIdx.x;
  if (i < n) dst[i] = (bf16_t)src[i];
}

// ---------------- pack W[256,256] f32 -> per-lane WMMA-B fragment order (bf16) ----------------
__global__ __launch_bounds__(256) void pack_w_k(const float* __restrict__ W,
                                                bf16_t* __restrict__ Wp) {
  int o = blockIdx.x * 256 + threadIdx.x;  // 65536
  int tile = o >> 9;
  int r = o & 511;
  int l = r >> 4, j = r & 15;
  int c = tile >> 3, kkIdx = tile & 7;
  int k = kkIdx * 32 + (l >> 4) * 16 + j;
  int n = c * 16 + (l & 15);
  Wp[o] = (bf16_t)W[k * DD + n];
}

// ---------------- GEMM: C[M,256] = A_bf16[M,256] @ W + bias ----------------
// grid (M/128, 16), block 256 (8 waves)
// TDM path: double-buffered 8KB A tiles (16KB dynamic LDS), DMA overlapped with WMMA
__global__ __launch_bounds__(256) void gemm_bf16_k(const bf16_t* __restrict__ A,
                                                   const bf16_t* __restrict__ Wp,
                                                   const float* __restrict__ bias,
                                                   float* __restrict__ Cf,
                                                   bf16_t* __restrict__ Cb) {
  extern __shared__ char smem[];
  bf16_t* sA = (bf16_t*)smem;  // 2 x (128*32) bf16 buffers
  int rowBase = blockIdx.x * 128;
  int colTile = blockIdx.y;
  int t = threadIdx.x, wave = t >> 5, lane = t & 31;
  v8f acc = {};
#if HAVE_TDM
  if (wave == 0) tdm_issue_tile(A, rowBase, 0, 0u);  // prologue: tile 0 -> buf 0
#endif
#pragma unroll
  for (int kkIdx = 0; kkIdx < 8; ++kkIdx) {
#if HAVE_TDM
    if (wave == 0) {
      if (kkIdx < 7) {
        // issue next tile into the other buffer, then wait for current (in-order)
        tdm_issue_tile(A, rowBase, (kkIdx + 1) * 32,
                       (unsigned)(((kkIdx + 1) & 1) * 8192));
        __builtin_amdgcn_s_wait_tensorcnt(1);
      } else {
        __builtin_amdgcn_s_wait_tensorcnt(0);
      }
    }
    __syncthreads();   // current tile visible to all waves
    const bf16_t* cur = sA + (kkIdx & 1) * 4096;
#else
    { // fallback: staged copy, 16 contiguous bf16 per thread
      int r = t >> 1, c = (t & 1) * 16;
      const uint4* src = (const uint4*)(A + (size_t)(rowBase + r) * DD + kkIdx * 32 + c);
      uint4* dst = (uint4*)(sA + r * 32 + c);
      dst[0] = src[0];
      dst[1] = src[1];
      if (kkIdx < 7)
        __builtin_prefetch(A + (size_t)(rowBase + r) * DD + kkIdx * 32 + 32 + c, 0, 3);
    }
    __syncthreads();
    const bf16_t* cur = sA;
#endif
    v16bf a = load_a_frag(cur + wave * 16 * 32, 32);
    v16bf b = load_bfrag_packed(Wp + ((size_t)colTile * 8 + kkIdx) * 512);
    acc = __builtin_amdgcn_wmma_f32_16x16x32_bf16(false, a, false, b, (short)0,
                                                  acc, false, false);
    __syncthreads();   // all waves done reading before buffer is re-targeted
  }
  int n = lane & 15, mb = (lane >> 4) * 8;
  int colBase = colTile * 16;
  float bb = bias ? bias[colBase + n] : 0.0f;
  int rowW = rowBase + wave * 16;
#pragma unroll
  for (int r = 0; r < 8; ++r) {
    float v = acc[r] + bb;
    size_t o = (size_t)(rowW + mb + r) * DD + colBase + n;
    if (Cf) Cf[o] = v;
    else    Cb[o] = (bf16_t)v;
  }
}

// ---------------- attention: per (b,h,row-block of 64) ----------------
// dynamic LDS layout (byte offsets; dynamic base == 0, no static LDS in kernel):
#define SQ_OFF    0u       // 64*32 bf16   = 4096
#define SK_OFF    4096u    // 256*32 bf16  = 16384   [key][hd]
#define SVT_OFF   20480u   // 32*256 bf16  = 16384   [hd][key] (transposed)
#define SS_OFF    36864u   // 64*256 f32   = 65536   (V staged here first)
#define SP_OFF    102400u  // 64*256 bf16  = 32768
#define ATTN_LDS  135168u

__global__ __launch_bounds__(256) void attn_k(const bf16_t* __restrict__ Q,
                                              const bf16_t* __restrict__ K,
                                              const bf16_t* __restrict__ V,
                                              const float* __restrict__ dist,
                                              const unsigned char* __restrict__ mask,
                                              bf16_t* __restrict__ AO) {
  extern __shared__ char smem[];
  bf16_t* sQ  = (bf16_t*)(smem + SQ_OFF);
  bf16_t* sK  = (bf16_t*)(smem + SK_OFF);
  bf16_t* sVt = (bf16_t*)(smem + SVT_OFF);
  float*  sS  = (float*) (smem + SS_OFF);
  bf16_t* sP  = (bf16_t*)(smem + SP_OFF);

  int blk = blockIdx.x;
  int mb = blk & 3;
  int h  = (blk >> 2) & 7;
  int b  = blk >> 5;
  int n0 = mb * 64;
  int t = threadIdx.x, wave = t >> 5, lane = t & 31;

  { // async stage Q (64x32), K (256x32), V staged into sS region
    int r = t >> 2, c = (t & 3) * 16;  // Q: one 16B chunk per thread
    async_ld_b128(SQ_OFF + (unsigned)(r * 64 + c),
                  (const char*)Q + (((size_t)b * NN + n0 + r) * DD + h * HD) * 2 + c);
#pragma unroll
    for (int q = 0; q < 4; ++q) {
      async_ld_b128(SK_OFF + (unsigned)(t * 64 + q * 16),
                    (const char*)K + (((size_t)b * NN + t) * DD + h * HD) * 2 + q * 16);
      async_ld_b128(SS_OFF + (unsigned)(t * 64 + q * 16),
                    (const char*)V + (((size_t)b * NN + t) * DD + h * HD) * 2 + q * 16);
    }
  }
  wait_async0();
  __syncthreads();

  { // transpose V: sVt[hd*256 + key] = Vstage[key*32 + hd]
    const bf16_t* vst = (const bf16_t*)(smem + SS_OFF);
#pragma unroll
    for (int c = 0; c < 32; ++c) sVt[c * 256 + t] = vst[t * 32 + c];
  }
  __syncthreads();

  // scores = QK^T * scale + dist, mask ; 64 tiles of 16x16, 8 per wave
  const float scale = 0.17677669529663687f; // 1/sqrt(32)
  for (int tile = wave; tile < 64; tile += 8) {
    int tr = tile & 3, tc = tile >> 2;
    v16bf a  = load_a_frag(sQ + tr * 16 * 32, 32);
    v16bf bf = load_bt_frag(sK + tc * 16 * 32, 32);  // contiguous 32B per lane
    v8f acc = {};
    acc = __builtin_amdgcn_wmma_f32_16x16x32_bf16(false, a, false, bf, (short)0,
                                                  acc, false, false);
    int nn2 = lane & 15, mbr = (lane >> 4) * 8;
    int gcol = tc * 16 + nn2;
    bool mk = mask[b * NN + gcol] != 0;
#pragma unroll
    for (int r = 0; r < 8; ++r) {
      int grow = tr * 16 + mbr + r;
      float s = acc[r] * scale + dist[((size_t)b * NN + n0 + grow) * NN + gcol];
      sS[grow * 256 + gcol] = mk ? s : -1e9f;
    }
  }
  __syncthreads();

  // softmax: 4 lanes per row
  {
    int row = t >> 2, sub = t & 3;
    float mx = -3.4e38f;
    for (int c = sub; c < 256; c += 4) mx = fmaxf(mx, sS[row * 256 + c]);
    mx = fmaxf(mx, __shfl_xor(mx, 1, 4));
    mx = fmaxf(mx, __shfl_xor(mx, 2, 4));
    float sum = 0.0f;
    for (int c = sub; c < 256; c += 4) {
      float e = __expf(sS[row * 256 + c] - mx);
      sS[row * 256 + c] = e;
      sum += e;
    }
    sum += __shfl_xor(sum, 1, 4);
    sum += __shfl_xor(sum, 2, 4);
    float inv = 1.0f / sum;
    for (int c = sub; c < 256; c += 4)
      sP[row * 256 + c] = (bf16_t)(sS[row * 256 + c] * inv);
  }
  __syncthreads();

  // out = P @ V : 8 tiles of 16x16 (4 row x 2 col), one per wave, K = 256
  {
    int tr = wave & 3, tc = wave >> 2;
    v8f o = {};
    for (int kk = 0; kk < 256; kk += 32) {
      v16bf a  = load_a_frag(sP + tr * 16 * 256 + kk, 256);
      v16bf bf = load_bt_frag(sVt + tc * 16 * 256 + kk, 256); // contiguous 32B
      o = __builtin_amdgcn_wmma_f32_16x16x32_bf16(false, a, false, bf, (short)0,
                                                  o, false, false);
    }
    int nn2 = lane & 15, mbr = (lane >> 4) * 8;
#pragma unroll
    for (int r = 0; r < 8; ++r)
      AO[((size_t)b * NN + n0 + tr * 16 + mbr + r) * DD + h * HD + tc * 16 + nn2] =
          (bf16_t)o[r];
  }
}

// ---------------- pooling score: one wave per row ----------------
__global__ __launch_bounds__(256) void pool_score_k(const float* __restrict__ enc,
                                                    const float* __restrict__ w,
                                                    float* __restrict__ score) {
  int row = blockIdx.x * 8 + (threadIdx.x >> 5);
  int lane = threadIdx.x & 31;
  float d = 0.0f, nw = 0.0f;
  for (int c = lane; c < DD; c += 32) {
    float wv = w[c];
    d += enc[(size_t)row * DD + c] * wv;
    nw += wv * wv;
  }
#pragma unroll
  for (int o = 16; o; o >>= 1) {
    d  += __shfl_xor(d, o, 32);
    nw += __shfl_xor(nw, o, 32);
  }
  if (lane == 0) score[row] = d / sqrtf(nw);
}

// ---------------- top-k per graph (bitonic sort of 256) ----------------
__global__ __launch_bounds__(256) void topk_k(const float* __restrict__ score,
                                              const float* __restrict__ enc,
                                              float* __restrict__ sub_x,
                                              long long* __restrict__ sub_batch,
                                              int* __restrict__ new_id) {
  __shared__ unsigned long long skey[256];
  __shared__ float sval[256];
  __shared__ int   spos[256];
  __shared__ int   sidx[128];
  __shared__ float stanh[128];
  int b = blockIdx.x, t = threadIdx.x;
  float sc = score[b * NN + t];
  sval[t] = sc;
  unsigned int f = __float_as_uint(sc);
  unsigned int m = (f & 0x80000000u) ? ~f : (f | 0x80000000u);
  skey[t] = ((unsigned long long)(~m) << 32) | (unsigned int)t; // asc = desc score, asc idx
  spos[t] = -1;
  __syncthreads();
  for (int k = 2; k <= 256; k <<= 1)
    for (int j = k >> 1; j > 0; j >>= 1) {
      int l = t ^ j;
      if (l > t) {
        unsigned long long a = skey[t], c = skey[l];
        bool asc = ((t & k) == 0);
        if ((a > c) == asc) { skey[t] = c; skey[l] = a; }
      }
      __syncthreads();
    }
  if (t < KSEL) {
    int idx = (int)(skey[t] & 0xffffffffu);
    sidx[t] = idx;
    stanh[t] = tanhf(sval[idx]);
    spos[idx] = t;
    sub_batch[b * KSEL + t] = (long long)b;
  }
  __syncthreads();
  new_id[b * NN + t] = (spos[t] >= 0) ? (b * KSEL + spos[t]) : SENT;
  for (int i = 0; i < KSEL; ++i) {
    int src = sidx[i];
    sub_x[((size_t)b * KSEL + i) * DD + t] =
        enc[((size_t)b * NN + src) * DD + t] * stanh[i];
  }
}

// ---------------- edge relabel -> sortable keys ----------------
__global__ __launch_bounds__(256) void edge_key_k(const long long* __restrict__ ei,
                                                  const int* __restrict__ new_id,
                                                  unsigned long long* __restrict__ keys) {
  int e = blockIdx.x * 256 + threadIdx.x;
  long long s = ei[e], d = ei[EE + e];
  int r = new_id[s], c = new_id[d];
  if (r >= SENT || c >= SENT) { r = SENT; c = SENT; }
  keys[e] = (unsigned long long)r * (unsigned long long)(SENT + 1) +
            (unsigned long long)c;
}

__global__ __launch_bounds__(256) void bitonic_pass_k(unsigned long long* __restrict__ keys,
                                                      int j, int k) {
  int i = blockIdx.x * 256 + threadIdx.x;
  int l = i ^ j;
  if (l > i) {
    unsigned long long a = keys[i], b = keys[l];
    bool asc = ((i & k) == 0);
    if ((a > b) == asc) { keys[i] = b; keys[l] = a; }
  }
}

// chained local passes for j <= 256 on 512-element LDS-resident windows
__global__ __launch_bounds__(256) void bitonic_local_k(unsigned long long* __restrict__ keys,
                                                       int k, int jstart) {
  __shared__ unsigned long long s[512];
  int base = blockIdx.x * 512, t = threadIdx.x;
  s[t] = keys[base + t];
  s[t + 256] = keys[base + t + 256];
  __syncthreads();
  for (int j = jstart; j > 0; j >>= 1) {
    int i = ((t & ~(j - 1)) << 1) | (t & (j - 1));
    int l = i | j;
    bool asc = (((base + i) & k) == 0);
    unsigned long long a = s[i], bb = s[l];
    if ((a > bb) == asc) { s[i] = bb; s[l] = a; }
    __syncthreads();
  }
  keys[base + t] = s[t];
  keys[base + t + 256] = s[t + 256];
}

__global__ __launch_bounds__(256) void edge_out_k(const unsigned long long* __restrict__ keys,
                                                  long long* __restrict__ sub_ei) {
  int e = blockIdx.x * 256 + threadIdx.x;
  unsigned long long key = keys[e];
  sub_ei[e]      = (long long)(key / (unsigned long long)(SENT + 1));
  sub_ei[EE + e] = (long long)(key % (unsigned long long)(SENT + 1));
}

// ---------------- launch ----------------
extern "C" void kernel_launch(void* const* d_in, const int* in_sizes, int n_in,
                              void* d_out, int out_size, void* d_ws, size_t ws_size,
                              hipStream_t stream) {
  const float* emb  = (const float*)d_in[0];
  const unsigned char* mask = (const unsigned char*)d_in[1];
  const float* dist = (const float*)d_in[2];
  const long long* edge_index = (const long long*)d_in[3];
  // d_in[4] = batch (derivable; unused)
  const float* Wq = (const float*)d_in[5];  const float* bq = (const float*)d_in[6];
  const float* Wk = (const float*)d_in[7];  const float* bk = (const float*)d_in[8];
  const float* Wv = (const float*)d_in[9];  const float* bv = (const float*)d_in[10];
  const float* Wo = (const float*)d_in[11]; const float* bo = (const float*)d_in[12];
  const float* pool_w = (const float*)d_in[13];

  // ---- workspace layout ----
  char* p = (char*)d_ws;
  const size_t actBytes = (size_t)MTOT * DD * sizeof(bf16_t); // 16 MB
  bf16_t* xb  = (bf16_t*)p; p += actBytes;
  bf16_t* wqp = (bf16_t*)p; p += (size_t)DD * DD * sizeof(bf16_t);
  bf16_t* wkp = (bf16_t*)p; p += (size_t)DD * DD * sizeof(bf16_t);
  bf16_t* wvp = (bf16_t*)p; p += (size_t)DD * DD * sizeof(bf16_t);
  bf16_t* wop = (bf16_t*)p; p += (size_t)DD * DD * sizeof(bf16_t);
  bf16_t* Qb  = (bf16_t*)p; p += actBytes;
  bf16_t* Kb  = (bf16_t*)p; p += actBytes;
  bf16_t* Vb  = (bf16_t*)p; p += actBytes;
  bf16_t* AOb = (bf16_t*)p; p += actBytes;
  float* scorep = (float*)p; p += (size_t)MTOT * sizeof(float);
  int* new_id   = (int*)p;   p += (size_t)MTOT * sizeof(int);
  unsigned long long* keys = (unsigned long long*)p; p += (size_t)EE * 8;

  // ---- output layout (tuple concatenated) ----
  float* enc_out = (float*)d_out;
  float* sub_x   = enc_out + (size_t)MTOT * DD;
  long long* sub_ei    = (long long*)(sub_x + (size_t)BB * KSEL * DD);
  long long* sub_batch = sub_ei + (size_t)2 * EE;

  // 1) conversions / weight fragment packing
  cvt_k<<<(MTOT * DD + 255) / 256, 256, 0, stream>>>(emb, xb, MTOT * DD);
  pack_w_k<<<DD * DD / 256, 256, 0, stream>>>(Wq, wqp);
  pack_w_k<<<DD * DD / 256, 256, 0, stream>>>(Wk, wkp);
  pack_w_k<<<DD * DD / 256, 256, 0, stream>>>(Wv, wvp);
  pack_w_k<<<DD * DD / 256, 256, 0, stream>>>(Wo, wop);

  // 2) Q/K/V projections (WMMA + double-buffered TDM, bf16 out)
  dim3 gg(MTOT / 128, DD / 16);
  const size_t gemmLds = 2 * 128 * 32 * sizeof(bf16_t); // 16KB (2 buffers)
  gemm_bf16_k<<<gg, 256, gemmLds, stream>>>(xb, wqp, bq, nullptr, Qb);
  gemm_bf16_k<<<gg, 256, gemmLds, stream>>>(xb, wkp, bk, nullptr, Kb);
  gemm_bf16_k<<<gg, 256, gemmLds, stream>>>(xb, wvp, bv, nullptr, Vb);

  // 3) attention (async-to-LDS staging + WMMA + LDS softmax)
  attn_k<<<BB * HH * (NN / 64), 256, ATTN_LDS, stream>>>(Qb, Kb, Vb, dist, mask, AOb);

  // 4) output projection -> encoder_result (f32)
  gemm_bf16_k<<<gg, 256, gemmLds, stream>>>(AOb, wop, bo, enc_out, nullptr);

  // 5) pooling scores + per-graph top-k + relabel
  pool_score_k<<<MTOT / 8, 256, 0, stream>>>(enc_out, pool_w, scorep);
  topk_k<<<BB, 256, 0, stream>>>(scorep, enc_out, sub_x, sub_batch, new_id);

  // 6) edge relabel -> keys -> hybrid bitonic sort -> decode
  edge_key_k<<<EE / 256, 256, 0, stream>>>(edge_index, new_id, keys);
  for (unsigned k = 2; k <= EE; k <<= 1) {
    unsigned j = k >> 1;
    for (; j > 256; j >>= 1)
      bitonic_pass_k<<<EE / 256, 256, 0, stream>>>(keys, (int)j, (int)k);
    bitonic_local_k<<<EE / 512, 256, 0, stream>>>(keys, (int)k, (int)j);
  }
  edge_out_k<<<EE / 256, 256, 0, stream>>>(keys, sub_ei);
}
